// NeuraLogic_84945863180634
// MI455X (gfx1250) — matrix operand order repaired and checked by reading, not compiled, not run on hardware
//
#include <hip/hip_runtime.h>
#include <hip/hip_bf16.h>

// GCN 3-layer + pooling for MI455X (gfx1250, wave32, WMMA f32_16x16x32_f16)

typedef __attribute__((ext_vector_type(16))) _Float16 v16h;
typedef __attribute__((ext_vector_type(8)))  float    v8f;

#define HDIM 64
#define FRAGS_PER_MAT 8              // (64/32 k-tiles) * (64/16 n-tiles)
#define FRAG_HALFS    (32 * 16)      // 32 lanes * 16 halves
#define MAT_HALFS     (FRAGS_PER_MAT * FRAG_HALFS)

// ---- WMMA fragment loaders (layouts per CDNA5 ISA 7.12.2) -------------------

// A fragment: 16x32 f16, rows row0..row0+15 of row-major f32 matrix X (ld=64),
// K-slice [k0,k0+32). Per lane the 16 elements are two contiguous 8-float runs
// -> 4 x b128 loads + packed f16 converts.
__device__ inline v16h load_a16x32(const float* __restrict__ X, int row0, int k0, int lane) {
  const int half = lane >> 4;
  const int m    = lane & 15;
  const float* xr = X + (size_t)(row0 + m) * HDIM + k0 + half * 8;  // 32B aligned
  float4 r0 = ((const float4*)xr)[0];
  float4 r1 = ((const float4*)xr)[1];
  float4 r2 = ((const float4*)(xr + 16))[0];
  float4 r3 = ((const float4*)(xr + 16))[1];
  v16h a;
  a[0]  = (_Float16)r0.x; a[1]  = (_Float16)r0.y; a[2]  = (_Float16)r0.z; a[3]  = (_Float16)r0.w;
  a[4]  = (_Float16)r1.x; a[5]  = (_Float16)r1.y; a[6]  = (_Float16)r1.z; a[7]  = (_Float16)r1.w;
  a[8]  = (_Float16)r2.x; a[9]  = (_Float16)r2.y; a[10] = (_Float16)r2.z; a[11] = (_Float16)r2.w;
  a[12] = (_Float16)r3.x; a[13] = (_Float16)r3.y; a[14] = (_Float16)r3.z; a[15] = (_Float16)r3.w;
  return a;
}

// B fragment built from row-major f32 W (slow path; used once in prep kernel).
// 32x16, K-slice [k0,k0+32), N-slice [n0,n0+16). Lanes 0-15: K=k0..k0+7,k0+16..k0+23,
// lanes 16-31: K=k0+8..k0+15,k0+24..k0+31.
__device__ inline v16h build_b32x16(const float* __restrict__ W, int k0, int n0, int lane) {
  const int kh = lane >> 4;
  const int n  = lane & 15;
  v16h b;
#pragma unroll
  for (int j = 0; j < 8; ++j) {
    int k = k0 + kh * 8 + 2 * j + (j >= 4 ? 8 : 0);
    b[2 * j]     = (_Float16)W[(size_t)k * HDIM + n0 + n];
    b[2 * j + 1] = (_Float16)W[(size_t)(k + 1) * HDIM + n0 + n];
  }
  return b;
}

// Fast B-fragment load from pre-converted fragment-layout f16 weights:
// one 32-byte per-lane load (2 x global_load_b128).
__device__ inline v16h load_bfrag(const _Float16* __restrict__ Wf, int k0, int n0, int lane) {
  const int frag = (k0 >> 5) * 4 + (n0 >> 4);
  return *(const v16h*)(Wf + ((size_t)frag * 32 + lane) * 16);
}

// ---- Kernel 0: convert one 64x64 f32 weight matrix -> f16 fragment layout ---
// <<<1, 256>>>: 8 waves = 8 fragments, each lane stores its 16 halves (32B).
__global__ __launch_bounds__(256) void prep_weights(const float* __restrict__ W,
                                                    _Float16* __restrict__ Wf) {
  const int wave = threadIdx.x >> 5;       // fragment id 0..7
  const int lane = threadIdx.x & 31;
  const int k0 = (wave >> 2) * 32;
  const int n0 = (wave & 3) * 16;
  v16h b = build_b32x16(W, k0, n0, lane);
  *(v16h*)(Wf + ((size_t)wave * 32 + lane) * 16) = b;
}

// ---- Kernel 1: H = X @ W  (X: nTiles*16 x 64, Wf: frag-layout f16) ----------
// One wave per 16-row tile; 4 16x16 accumulators cover all 64 output columns.
__global__ __launch_bounds__(256) void gemm_xw(const float* __restrict__ X,
                                               const _Float16* __restrict__ Wf,
                                               float* __restrict__ Hout, int nTiles) {
  const int wave = threadIdx.x >> 5;
  const int lane = threadIdx.x & 31;
  const int tile = blockIdx.x * (blockDim.x >> 5) + wave;
  if (tile >= nTiles) return;               // wave-uniform: EXEC stays all-ones
  const int row0 = tile * 16;

  v8f c0 = {}, c1 = {}, c2 = {}, c3 = {};
#pragma unroll
  for (int k0 = 0; k0 < HDIM; k0 += 32) {
    v16h a  = load_a16x32(X, row0, k0, lane);
    v16h b0 = load_bfrag(Wf, k0,  0, lane);
    v16h b1 = load_bfrag(Wf, k0, 16, lane);
    v16h b2 = load_bfrag(Wf, k0, 32, lane);
    v16h b3 = load_bfrag(Wf, k0, 48, lane);
    c0 = __builtin_amdgcn_wmma_f32_16x16x32_f16(false, a, false, b0, (short)0, c0, false, false);
    c1 = __builtin_amdgcn_wmma_f32_16x16x32_f16(false, a, false, b1, (short)0, c1, false, false);
    c2 = __builtin_amdgcn_wmma_f32_16x16x32_f16(false, a, false, b2, (short)0, c2, false, false);
    c3 = __builtin_amdgcn_wmma_f32_16x16x32_f16(false, a, false, b3, (short)0, c3, false, false);
  }

  const int half = lane >> 4;
  const int n    = lane & 15;
#pragma unroll
  for (int r = 0; r < 8; ++r) {
    float* hr = Hout + (size_t)(row0 + r + half * 8) * HDIM + n;
    hr[0]  = c0[r];
    hr[16] = c1[r];
    hr[32] = c2[r];
    hr[48] = c3[r];
  }
}

// ---- Kernel 2: edge scatter  acc[dst] += h[src]  ----------------------------
// One thread per (edge, feature-pair): float2 coalesced reads, f32 atomics into
// an L2-resident (12.8 MB) accumulator.
__global__ __launch_bounds__(256) void scatter_edges(const float* __restrict__ h,
                                                     const int* __restrict__ src,
                                                     const int* __restrict__ dst,
                                                     float* __restrict__ acc, int nEdges) {
  long long t = (long long)blockIdx.x * blockDim.x + threadIdx.x;
  int e = (int)(t >> 5);                    // 32 threads per edge, 2 feats each
  int f = (int)(t & 31) * 2;
  if (e >= nEdges) return;
  int s = src[e];
  int d = dst[e];
  float2 v = *(const float2*)(h + (size_t)s * HDIM + f);
  float* ap = acc + (size_t)d * HDIM + f;
  atomicAdd(ap,     v.x);
  atomicAdd(ap + 1, v.y);
}

// ---- Kernel 3: relu + segmented pool sum ------------------------------------
// batch[] is sorted, so each feature-lane run-length accumulates and flushes
// one atomic per graph boundary instead of one per node.
__global__ __launch_bounds__(64) void relu_pool(const float* __restrict__ acc,
                                                float* __restrict__ xout,
                                                const int* __restrict__ batch,
                                                float* __restrict__ pool, int nNodes) {
  const int f  = threadIdx.x;               // 0..63
  const int n0 = blockIdx.x * 128;
  const int nEnd = (n0 + 128 < nNodes) ? (n0 + 128) : nNodes;
  float run = 0.0f;
  int curg = -1;
  for (int nd = n0; nd < nEnd; ++nd) {
    int g = batch[nd];
    float v = acc[(size_t)nd * HDIM + f];
    v = v > 0.0f ? v : 0.0f;
    xout[(size_t)nd * HDIM + f] = v;
    if (g != curg) {
      if (curg >= 0) atomicAdd(&pool[(size_t)curg * HDIM + f], run);
      run = 0.0f;
      curg = g;
    }
    run += v;
  }
  if (curg >= 0) atomicAdd(&pool[(size_t)curg * HDIM + f], run);
}

// ---- Kernel 4: Out = relu(P1@Wp1 + P2@Wp2 + P3@Wp3), 256x64 -----------------
__global__ __launch_bounds__(256) void pool_gemm(const float* __restrict__ P1,
                                                 const float* __restrict__ P2,
                                                 const float* __restrict__ P3,
                                                 const _Float16* __restrict__ Wf1,
                                                 const _Float16* __restrict__ Wf2,
                                                 const _Float16* __restrict__ Wf3,
                                                 float* __restrict__ Out, int nRowTiles) {
  const int wave = threadIdx.x >> 5;
  const int lane = threadIdx.x & 31;
  const int tid  = blockIdx.x * (blockDim.x >> 5) + wave;
  if (tid >= nRowTiles * 4) return;
  const int row0 = (tid >> 2) * 16;
  const int n0   = (tid & 3) * 16;

  v8f c = {};
#pragma unroll
  for (int k0 = 0; k0 < HDIM; k0 += 32) {
    v16h a1 = load_a16x32(P1, row0, k0, lane);
    v16h b1 = load_bfrag(Wf1, k0, n0, lane);
    c = __builtin_amdgcn_wmma_f32_16x16x32_f16(false, a1, false, b1, (short)0, c, false, false);
    v16h a2 = load_a16x32(P2, row0, k0, lane);
    v16h b2 = load_bfrag(Wf2, k0, n0, lane);
    c = __builtin_amdgcn_wmma_f32_16x16x32_f16(false, a2, false, b2, (short)0, c, false, false);
    v16h a3 = load_a16x32(P3, row0, k0, lane);
    v16h b3 = load_bfrag(Wf3, k0, n0, lane);
    c = __builtin_amdgcn_wmma_f32_16x16x32_f16(false, a3, false, b3, (short)0, c, false, false);
  }

  const int half = lane >> 4;
  const int n    = lane & 15;
#pragma unroll
  for (int r = 0; r < 8; ++r) {
    float v = c[r];
    Out[(size_t)(row0 + r + half * 8) * HDIM + n0 + n] = v > 0.0f ? v : 0.0f;
  }
}

// ---- Host side --------------------------------------------------------------

extern "C" void kernel_launch(void* const* d_in, const int* in_sizes, int n_in,
                              void* d_out, int out_size, void* d_ws, size_t ws_size,
                              hipStream_t stream) {
  const float* x     = (const float*)d_in[0];
  const int*   ei    = (const int*)d_in[1];     // [2, E] flat: row0=src, row1=dst
  const int*   batch = (const int*)d_in[2];
  const float* Wmat[6] = {(const float*)d_in[3], (const float*)d_in[4],
                          (const float*)d_in[5], (const float*)d_in[6],
                          (const float*)d_in[7], (const float*)d_in[8]};

  const int nNodes  = in_sizes[0] / HDIM;       // 50000
  const int nEdges  = in_sizes[1] / 2;          // 1000000
  const int nGraphs = out_size / HDIM;          // 256
  const int* src = ei;
  const int* dst = ei + nEdges;

  // Workspace layout (node buffers are 256B-aligned: nodeBytes = 12,800,000)
  char*  ws        = (char*)d_ws;
  size_t nodeBytes = (size_t)nNodes * HDIM * sizeof(float);
  float*     h    = (float*)(ws);
  float*     acc  = (float*)(ws + nodeBytes);
  float*     xcur = (float*)(ws + 2 * nodeBytes);
  float*     pool = (float*)(ws + 3 * nodeBytes);            // 3 x [nGraphs, 64]
  size_t     poolElems = (size_t)nGraphs * HDIM;
  _Float16*  Wf   = (_Float16*)(ws + 3 * nodeBytes + 3 * poolElems * sizeof(float));
                                                              // 6 x 8KB frag-layout f16

  // One-time-per-call prep: weights -> f16 fragment layout; zero pool sums.
  for (int m = 0; m < 6; ++m)
    prep_weights<<<1, 256, 0, stream>>>(Wmat[m], Wf + (size_t)m * MAT_HALFS);
  hipMemsetAsync(pool, 0, 3 * poolElems * sizeof(float), stream);

  const int nTiles = nNodes / 16;               // 3125 (exact)
  const long long edgeThreads = (long long)nEdges * (HDIM / 2);
  const int scatterBlocks = (int)((edgeThreads + 255) / 256);

  const float* xin = x;
  for (int l = 0; l < 3; ++l) {
    hipMemsetAsync(acc, 0, nodeBytes, stream);
    gemm_xw<<<(nTiles + 7) / 8, 256, 0, stream>>>(xin, Wf + (size_t)l * MAT_HALFS, h, nTiles);
    scatter_edges<<<scatterBlocks, 256, 0, stream>>>(h, src, dst, acc, nEdges);
    relu_pool<<<(nNodes + 127) / 128, 64, 0, stream>>>(acc, xcur, batch,
                                                       pool + (size_t)l * poolElems, nNodes);
    xin = xcur;
  }

  const int gTiles = nGraphs / 16;              // 16
  pool_gemm<<<(gTiles * 4 + 7) / 8, 256, 0, stream>>>(
      pool, pool + poolElems, pool + 2 * poolElems,
      Wf + 3 * (size_t)MAT_HALFS, Wf + 4 * (size_t)MAT_HALFS, Wf + 5 * (size_t)MAT_HALFS,
      (float*)d_out, gTiles);
}
